// GDKR_48120813584731
// MI455X (gfx1250) — matrix operand order repaired and checked by reading, not compiled, not run on hardware
//
#include <hip/hip_runtime.h>
#include <math.h>

typedef __attribute__((ext_vector_type(16))) _Float16 v16h;
typedef __attribute__((ext_vector_type(8)))  float    v8f;
typedef __attribute__((ext_vector_type(4)))  float    f4;

#define TWO_PI      6.28318530717958647692f
#define INV_SQRT_F  0.0625f
#define LR_DECAY    0.977f
#define MOMENTUM    0.85f

#define NFUNC 256
#define NTGT  1024
#define MEMSZ 65536

// ---- output layout (float offsets), matches reference concatenation order ----
#define O_AMP   67108864
#define O_PHASE 67371008
#define O_FREQ  67633152
#define O_BIAS  67633408
#define O_AG    67634432
#define O_PG    67896576
#define O_FG    68158720
#define O_LR    68158976

// ---- workspace layout (float offsets) ----
#define WS_FR     0        // 256  fr[f]
#define WS_DFR    256      // 256  d(fr)/d(freq)
#define WS_TB     512      // 16   t_b (float) for current iteration
#define WS_TS     544      // 32   gather indices (stored as int)
#define WS_SCAL   576      // 2    [0]=||ag||^2  [1]=||pg||^2
#define WS_FGRAW  640      // 256  sum_k contributions to freq grad
#define WS_SINA   1024     // 16x256 sin(2*pi*t_b*fr)
#define WS_COSA   5120     // 16x256 cos(2*pi*t_b*fr)
#define WS_G      9216     // 16x1024 f16  G = -(1/16)*sign(zb - zhat)
#define WS_TG     13312    // 16x1024 f16  t_b * G
#define WS_AG     17408    // 256x1024 amp grad (data+reg)
#define WS_PG     279552   // 256x1024 phase grad
// total ~2.07 MB

__device__ __forceinline__ v8f wmma32(v16h a, v16h b, v8f c) {
  // emits v_wmma_f32_16x16x32_f16
  return __builtin_amdgcn_wmma_f32_16x16x32_f16(false, a, false, b, (short)0, c, false, false);
}

// sin(2*pi*u), cos(2*pi*u) via the hardware trig units (argument in revolutions).
// frac-reduction keeps the argument in the valid hardware domain.
__device__ __forceinline__ void sincos_rev(float u, float* s, float* c) {
  float uf = u - floorf(u);
  *s = __builtin_amdgcn_sinf(uf);
  *c = __builtin_amdgcn_cosf(uf);
}

__device__ __forceinline__ float wave_sum32(float v) {
#pragma unroll
  for (int m = 1; m < 32; m <<= 1) v += __shfl_xor(v, m, 32);
  return v;
}
__device__ __forceinline__ float half_sum16(float v) {
#pragma unroll
  for (int m = 1; m < 16; m <<= 1) v += __shfl_xor(v, m, 32);
  return v;
}

// ---------- 256 MB streaming copy: memory -> out[0..], NT to spare L2 ----------
__global__ void k_copy(const f4* __restrict__ src, f4* __restrict__ dst, int n4) {
  int stride = gridDim.x * blockDim.x;
  for (int i = blockIdx.x * blockDim.x + threadIdx.x; i < n4; i += stride) {
    f4 v = __builtin_nontemporal_load(src + i);
    __builtin_nontemporal_store(v, dst + i);
  }
}

// ---------- init: row substitute, param copies, bias/lr update, ts indices ----------
__global__ void k_init(const float* __restrict__ z, const int* __restrict__ t_memory,
                       const float* __restrict__ amp, const float* __restrict__ phase,
                       const float* __restrict__ freq, const float* __restrict__ bias,
                       const float* __restrict__ ampg, const float* __restrict__ phg,
                       const float* __restrict__ fgr, const float* __restrict__ wedge,
                       const float* __restrict__ lr, const int* __restrict__ t,
                       const int* __restrict__ dsz, float* __restrict__ out,
                       float* __restrict__ ws) {
  int tid = blockIdx.x * blockDim.x + threadIdx.x;
  int stride = gridDim.x * blockDim.x;
  int ds = dsz[0];
  int wi = ds % MEMSZ;
  for (int e = tid; e < NFUNC * NTGT; e += stride) {
    out[O_AMP + e]   = amp[e];
    out[O_PHASE + e] = phase[e];
    out[O_AG + e]    = ampg[e];
    out[O_PG + e]    = phg[e];
  }
  if (tid < NTGT) {
    out[wi * NTGT + tid] = z[tid];                       // memory.at[write_idx].set(z)
    float nn = (float)(ds + 1);                          // n = data_size after increment
    out[O_BIAS + tid] = bias[tid] * (nn / (nn + 1.f)) + z[tid] * (1.f / (nn + 1.f));
  }
  if (tid < NFUNC) {
    out[O_FREQ + tid] = freq[tid];
    out[O_FG + tid]   = fgr[tid];
  }
  if (tid < 32) {
    float w = fabsf(wedge[tid] + wedge[32 + tid] - 1.f); // |wedge_r.sum(0) - 1|
    int mi = ds < MEMSZ ? ds : MEMSZ;
    int tsv = ds - (int)(w * (float)mi);                 // i = data_size-1 (orig ds)
    tsv %= MEMSZ; if (tsv < 0) tsv += MEMSZ;
    ((int*)ws)[WS_TS + tid] = tsv;
  }
  if (tid == 0) out[O_LR] = lr[0] * LR_DECAY;
}

// ---------- per-iteration prep: fr/dfr, t_b gather, SinA/CosA, zero accumulators ----------
__global__ void k_prep(const int* __restrict__ t_memory, const int* __restrict__ t,
                       const int* __restrict__ dsz, const float* __restrict__ cfreq,
                       float* __restrict__ out, float* __restrict__ ws, int iter) {
  int f = threadIdx.x;  // 1 block x 256 threads
  float fv = out[O_FREQ + f];
  float th = tanhf(fv);
  float x  = cfreq[f] + th * (64.0f / 256.0f);
  float s  = 1.f / (1.f + expf(-x));
  ws[WS_FR + f]  = s * 0.5f;
  ws[WS_DFR + f] = 0.5f * s * (1.f - s) * (1.f - th * th) * 0.25f;
  ws[WS_FGRAW + f] = 0.f;
  if (f < 2) ws[WS_SCAL + f] = 0.f;
  if (f < 16) {
    int idx = ((const int*)ws)[WS_TS + iter * 16 + f];
    int wi  = dsz[0] % MEMSZ;
    ws[WS_TB + f] = (idx == wi) ? (float)t[0] : (float)t_memory[idx];
  }
  __syncthreads();
  for (int e = f; e < 16 * NFUNC; e += 256) {
    int b = e >> 8, ff = e & 255;
    float u = ws[WS_TB + b] * ws[WS_FR + ff];  // revolutions
    float sa, ca; sincos_rev(u, &sa, &ca);
    ws[WS_SINA + e] = sa;
    ws[WS_COSA + e] = ca;
  }
}

// ---------- forward: zhat = (SinA@(cosp*amp) + CosA@(sinp*amp))/16 + bias ; G, tG ----------
// 64 waves, one 16x16 (batch x target) tile each; K loop over 256 in chunks of 32.
__global__ void k_forward(const float* __restrict__ memory, const float* __restrict__ z,
                          const int* __restrict__ dsz, float* __restrict__ out,
                          float* __restrict__ ws, int iter) {
  const int tid  = blockIdx.x * blockDim.x + threadIdx.x;
  const int wid  = tid >> 5;
  const int lane = threadIdx.x & 31;
  const int l    = lane & 15;
  const int hi   = (lane >= 16) ? 1 : 0;
  const int n    = wid * 16 + l;
  const float* SinA = ws + WS_SINA;
  const float* CosA = ws + WS_COSA;
  v8f acc = {};
  for (int kc = 0; kc < NFUNC; kc += 32) {
    v16h aS, aC, bC, bS;
#pragma unroll
    for (int h = 0; h < 16; ++h) {
      // A 16x32 f16 layout: lanes0-15 K={0..7,16..23}, lanes16-31 K={8..15,24..31}
      int kA = kc + ((h < 8) ? h : h + 8) + (hi ? 8 : 0);
      aS[h] = (_Float16)SinA[l * NFUNC + kA];
      aC[h] = (_Float16)CosA[l * NFUNC + kA];
      // B 32x16 f16 layout: lanes0-15 K={0..7,16..23}, lanes16-31 K={8..15,24..31}, N=lane%16
      int kB = kc + ((h < 8) ? h : h + 8) + (hi ? 8 : 0);
      float ph = out[O_PHASE + kB * NTGT + n];
      float am = out[O_AMP + kB * NTGT + n];
      float sp, cp; sincos_rev(ph, &sp, &cp);
      bC[h] = (_Float16)(cp * am);
      bS[h] = (_Float16)(sp * am);
    }
    acc = wmma32(aS, bC, acc);
    acc = wmma32(aC, bS, acc);
  }
  const int wi = dsz[0] % MEMSZ;
  const int* ts = (const int*)ws + WS_TS;
  _Float16* G  = (_Float16*)(ws + WS_G);
  _Float16* TG = (_Float16*)(ws + WS_TG);
  float bv = out[O_BIAS + n];
#pragma unroll
  for (int v = 0; v < 8; ++v) {
    int m = v + (hi ? 8 : 0);                 // C/D: VGPR v -> M=v (+8 upper lanes)
    int row = ts[iter * 16 + m];
    float zb = (row == wi) ? z[n] : memory[row * NTGT + n];
    float zh = acc[v] * INV_SQRT_F + bv;
    float d  = zb - zh;
    float g  = (d > 0.f) ? -0.0625f : ((d < 0.f) ? 0.0625f : 0.f);  // -(1/B)*sign
    G[m * NTGT + n]  = (_Float16)g;
    TG[m * NTGT + n] = (_Float16)(ws[WS_TB + m] * g);
  }
}

// ---------- backward: U,V,Ut,Vt via 4 WMMAs (K=16 zero-padded); ag/pg/fg + norms ----------
// 1024 waves: f-tile = wid/64, n-tile = wid%64.
__global__ void k_backward(float* __restrict__ out, float* __restrict__ ws, int iter) {
  const int tid  = blockIdx.x * blockDim.x + threadIdx.x;
  const int wid  = tid >> 5;
  const int lane = threadIdx.x & 31;
  const int l    = lane & 15;
  const int hi   = (lane >= 16) ? 1 : 0;
  const int f0   = (wid >> 6) * 16;
  const int n    = (wid & 63) * 16 + l;
  const float* SinA = ws + WS_SINA;
  const float* CosA = ws + WS_COSA;
  const _Float16* G  = (const _Float16*)(ws + WS_G);
  const _Float16* TG = (const _Float16*)(ws + WS_TG);
  v16h aS = {}, aC = {}, bG = {}, bT = {};
#pragma unroll
  for (int h = 0; h < 16; ++h) {
    int kA = ((h < 8) ? h : h + 8) + (hi ? 8 : 0);   // valid K (= batch b) < 16
    if (kA < 16) {
      aS[h] = (_Float16)SinA[kA * NFUNC + (f0 + l)]; // A = SinA^T : A[f][b]
      aC[h] = (_Float16)CosA[kA * NFUNC + (f0 + l)];
    }
    int kB = ((h < 8) ? h : h + 8) + (hi ? 8 : 0);
    if (kB < 16) {
      bG[h] = G[kB * NTGT + n];
      bT[h] = TG[kB * NTGT + n];
    }
  }
  v8f zc = {};
  v8f U  = wmma32(aS, bG, zc);
  v8f V  = wmma32(aC, bG, zc);
  v8f Ut = wmma32(aS, bT, zc);
  v8f Vt = wmma32(aC, bT, zc);
  float a2 = 0.f, p2 = 0.f;
#pragma unroll
  for (int v = 0; v < 8; ++v) {
    int f = f0 + v + (hi ? 8 : 0);
    int e = f * NTGT + n;
    float ph = out[O_PHASE + e];
    float am = out[O_AMP + e];
    float sp, cp; sincos_rev(ph, &sp, &cp);
    float ag = INV_SQRT_F * (cp * U[v] + sp * V[v]);           // data term (G has -(1/B))
    if (am != 0.f) ag += 0.005f * copysignf(rsqrtf(fabsf(am)), am);  // reg grad
    float pg  = (TWO_PI * INV_SQRT_F) * am * (cp * V[v]  - sp * U[v]);
    float fgc = (TWO_PI * INV_SQRT_F) * am * (cp * Vt[v] - sp * Ut[v]);
    ws[WS_AG + e] = ag;
    ws[WS_PG + e] = pg;
    a2 += ag * ag;
    p2 += pg * pg;
    float s = half_sum16(fgc);                                  // sum over 16 n-columns
    if (l == 0) atomicAdd(&ws[WS_FGRAW + f], s);
  }
  a2 = wave_sum32(a2);
  p2 = wave_sum32(p2);
  if (lane == 0) {
    atomicAdd(&ws[WS_SCAL + 0], a2);
    atomicAdd(&ws[WS_SCAL + 1], p2);
  }
}

// ---------- freq update (256 elems, one block) ----------
__global__ void k_upfreq(const float* __restrict__ lr, float* __restrict__ out,
                         float* __restrict__ ws) {
  __shared__ float red[NFUNC];
  int f = threadIdx.x;
  float fg = ws[WS_FGRAW + f] * ws[WS_DFR + f];
  red[f] = fg * fg;
  __syncthreads();
  for (int s = NFUNC / 2; s > 0; s >>= 1) {
    if (f < s) red[f] += red[f + s];
    __syncthreads();
  }
  float norm = sqrtf(red[0]);
  float lr1  = lr[0] * LR_DECAY;
  float g = out[O_FG + f] * MOMENTUM + fg / (norm + 1.f);
  out[O_FG + f] = g;
  out[O_FREQ + f] -= g * lr1;
}

// ---------- amp/phase update (262144 elems) ----------
__global__ void k_upap(const float* __restrict__ lr, float* __restrict__ out,
                       float* __restrict__ ws) {
  float na  = sqrtf(ws[WS_SCAL + 0]);
  float npn = sqrtf(ws[WS_SCAL + 1]);
  float lr1 = lr[0] * LR_DECAY;
  int stride = gridDim.x * blockDim.x;
  for (int e = blockIdx.x * blockDim.x + threadIdx.x; e < NFUNC * NTGT; e += stride) {
    float ag = ws[WS_AG + e] / (na + 1.f);
    float g  = out[O_AG + e] * MOMENTUM + ag;
    out[O_AG + e] = g;
    out[O_AMP + e] -= g * lr1;
    float pg = ws[WS_PG + e] / (npn + 1.f);
    float g2 = out[O_PG + e] * MOMENTUM + pg;
    out[O_PG + e] = g2;
    float ph = out[O_PHASE + e] - g2 * lr1;
    ph -= floorf(ph);                         // jnp.mod(x, 1.0)
    out[O_PHASE + e] = ph;
  }
}

extern "C" void kernel_launch(void* const* d_in, const int* in_sizes, int n_in,
                              void* d_out, int out_size, void* d_ws, size_t ws_size,
                              hipStream_t stream) {
  const float* z      = (const float*)d_in[0];
  const float* memory = (const float*)d_in[1];
  const int*   t_mem  = (const int*)d_in[2];
  const float* amp    = (const float*)d_in[3];
  const float* phase  = (const float*)d_in[4];
  const float* freq   = (const float*)d_in[5];
  const float* bias   = (const float*)d_in[6];
  const float* ampg   = (const float*)d_in[7];
  const float* phg    = (const float*)d_in[8];
  const float* fgr    = (const float*)d_in[9];
  const float* cfreq  = (const float*)d_in[10];
  const float* wedge  = (const float*)d_in[11];
  const float* lr     = (const float*)d_in[12];
  const int*   t      = (const int*)d_in[13];
  const int*   dsz    = (const int*)d_in[14];
  float* out = (float*)d_out;
  float* ws  = (float*)d_ws;

  (void)in_sizes; (void)n_in; (void)out_size; (void)ws_size;

  // 256 MB streaming copy (HBM-bound critical path)
  k_copy<<<8192, 256, 0, stream>>>((const f4*)memory, (f4*)out, (MEMSZ * NTGT) / 4);
  // row substitute + param init + bias/lr/ts
  k_init<<<1024, 256, 0, stream>>>(z, t_mem, amp, phase, freq, bias, ampg, phg, fgr,
                                   wedge, lr, t, dsz, out, ws);
  for (int iter = 0; iter < 2; ++iter) {
    k_prep<<<1, 256, 0, stream>>>(t_mem, t, dsz, cfreq, out, ws, iter);
    k_forward<<<8, 256, 0, stream>>>(memory, z, dsz, out, ws, iter);
    k_backward<<<128, 256, 0, stream>>>(out, ws, iter);
    k_upfreq<<<1, 256, 0, stream>>>(lr, out, ws);
    k_upap<<<256, 256, 0, stream>>>(lr, out, ws);
  }
}
